// Mamba_Block_35897336660104
// MI455X (gfx1250) — compile-verified, matrix operand order checked by reading, not run on hardware
//
#include <hip/hip_runtime.h>

#define D_MODEL 1024
#define D_STATE 16
#define D_CONV 4
#define DEPTH 5
#define D_INNER 2048
#define DT_RANK 64
#define BATCH 2
#define SEQ 1024
#define MROWS (BATCH * SEQ)            /* 2048 */
#define XDBL_W (DT_RANK + 2 * D_STATE) /* 96 */

typedef __attribute__((ext_vector_type(16))) __bf16 bf16x16;
typedef __attribute__((ext_vector_type(8))) float f32x8;
typedef __attribute__((ext_vector_type(4))) unsigned int u32x4;
typedef __attribute__((ext_vector_type(8))) int i32x8;
typedef __attribute__((ext_vector_type(4))) int i32x4;

// round-half-up f32 -> bf16 (cheap: 1 add + shift / perm)
__device__ __forceinline__ unsigned short bf16_of(float f) {
  return (unsigned short)((__builtin_bit_cast(unsigned int, f) + 0x8000u) >> 16);
}
// pack two f32 into bf16x2 via v_perm_b32 (dst = {hi16(b), hi16(a)})
__device__ __forceinline__ unsigned int pack_bf16(float a, float b) {
  unsigned int ua = __builtin_bit_cast(unsigned int, a) + 0x8000u;
  unsigned int ub = __builtin_bit_cast(unsigned int, b) + 0x8000u;
  return __builtin_amdgcn_perm(ub, ua, 0x07060302u);
}

union FragBF {
  bf16x16 v;
  uint4 q[2];
};

// ---------------------------------------------------------------------------
// TDM 2D tile load: global (bf16, row-major, row stride = `stride` elems)
//   -> LDS at byte offset lds_off, rows padded 64B->80B (pad_interval=16 dw,
//      pad_amount=4 dw) for conflict-free ds_load_b128 fragment reads.
// D# built per ISA 8.3/8.4 (group0 128b, group1 256b; groups 2/3 zero => 2D).
// This toolchain exposes the 6-arg builtin form.
// ---------------------------------------------------------------------------
__device__ __forceinline__ void tdm_load_2d(unsigned int lds_off,
                                            const void* gaddr,
                                            unsigned int tile_cols,
                                            unsigned int tile_rows,
                                            unsigned int tensor_cols,
                                            unsigned int tensor_rows,
                                            unsigned int stride_elems) {
  unsigned long long ga = (unsigned long long)gaddr;
  u32x4 g0;
  g0[0] = 1u;                                   // count=1, user mode, no gather
  g0[1] = lds_off;                              // lds_addr [63:32]
  g0[2] = (unsigned int)(ga & 0xFFFFFFFFu);     // global_addr[31:0]
  g0[3] = (unsigned int)((ga >> 32) & 0x1FFFFFFu) | 0x80000000u; // addr hi | type=2
  i32x8 g1;
  g1[0] = (int)((1u << 16)    // data_size = 2 bytes
                | (1u << 20)  // pad_enable
                | (3u << 22)  // pad_interval: 16 DWORDs (one 64B tile row)
                | (3u << 25)); // pad_amount: 4 DWORDs (16B)
  g1[1] = (int)(tensor_cols << 16);                              // dim0[15:0]
  g1[2] = (int)((tensor_cols >> 16) | (tensor_rows << 16));      // dim0 hi|dim1 lo
  g1[3] = (int)((tensor_rows >> 16) | (tile_cols << 16));        // dim1 hi|tile0
  g1[4] = (int)(tile_rows & 0xFFFFu);                            // tile1, tile2=0
  g1[5] = (int)stride_elems;                                     // dim0_stride lo
  g1[6] = 0;                                                     // stride hi
  g1[7] = 0;
  i32x4 z4 = {0, 0, 0, 0};
  i32x8 z8 = {0, 0, 0, 0, 0, 0, 0, 0};
  __builtin_amdgcn_tensor_load_to_lds(g0, g1, z4, z4, z8, 0);
}

// ---------------------------------------------------------------------------
// WMMA GEMM: C[M,N] = epilogue( A[M,K](bf16) * W[N,K](bf16)^T )
//   EPI 0: f32 store            EPI 1: softplus(C + bias[n])
//   EPI 2: C + resid (f32) and bf16 copy    EPI 3: f32 store and bf16 copy
// WG tile 128x256, BK=32; 8 waves as 2x4, each wave 64x64 = 4x4 WMMA frags.
// A/B tiles staged by TDM (double buffered), s_wait_tensorcnt + barrier sync.
// ---------------------------------------------------------------------------
constexpr int BM = 128, BN = 256, BK = 32;
constexpr int LDSW = 40; // 80B padded row stride in ushorts

template <int EPI>
__global__ __launch_bounds__(256) void gemm_wmma_bf16(
    const unsigned short* __restrict__ A, int lda,
    const unsigned short* __restrict__ W, int ldw, float* __restrict__ C,
    int ldc, unsigned short* __restrict__ Cbf, const float* __restrict__ bias,
    const float* __restrict__ resid, int ldr, int M, int N, int K) {
  __shared__ __align__(128) unsigned short sA[2][BM][LDSW];
  __shared__ __align__(128) unsigned short sB[2][BN][LDSW];

  const int tid = threadIdx.x;
  const int lane = tid & 31;
  const int wave = tid >> 5;
  const int wmBase = (wave >> 2) * 64; // 2 waves in M
  const int wnBase = (wave & 3) * 64;  // 4 waves in N
  const int bm = blockIdx.x * BM;
  const int bn = blockIdx.y * BN;
  const int l15 = lane & 15;
  const int khalf = lane >> 4; // selects K halves per ISA A/B layout

  const unsigned int ldsA0 = (unsigned int)(size_t)&sA[0][0][0];
  const unsigned int ldsA1 = (unsigned int)(size_t)&sA[1][0][0];
  const unsigned int ldsB0 = (unsigned int)(size_t)&sB[0][0][0];
  const unsigned int ldsB1 = (unsigned int)(size_t)&sB[1][0][0];

  const unsigned int bRows = ((unsigned int)(N - bn) < (unsigned int)BN)
                                 ? (unsigned int)(N - bn)
                                 : (unsigned int)BN; // TDM zero-fills the rest

  const int nk = K / BK;
  f32x8 acc[4][4] = {};

  if (wave == 0) { // one TDM issue per workgroup (wave-level op)
    tdm_load_2d(ldsA0, A + (size_t)bm * lda, BK, BM, BK, BM, (unsigned)lda);
    tdm_load_2d(ldsB0, W + (size_t)bn * ldw, BK, BN, BK, bRows, (unsigned)ldw);
  }

  for (int kt = 0; kt < nk; ++kt) {
    if (wave == 0) {
      if (kt + 1 < nk) { // prefetch next tile pair into other buffer
        const size_t ko = (size_t)(kt + 1) * BK;
        tdm_load_2d((kt & 1) ? ldsA0 : ldsA1, A + (size_t)bm * lda + ko, BK, BM,
                    BK, BM, (unsigned)lda);
        tdm_load_2d((kt & 1) ? ldsB0 : ldsB1, W + (size_t)bn * ldw + ko, BK, BN,
                    BK, bRows, (unsigned)ldw);
        __builtin_amdgcn_s_wait_tensorcnt((short)2); // current pair done
      } else {
        __builtin_amdgcn_s_wait_tensorcnt((short)0);
      }
    }
    __syncthreads();

    const unsigned short(*tA)[LDSW] = sA[kt & 1];
    const unsigned short(*tB)[LDSW] = sB[kt & 1];
    FragBF af[4], bfr[4];
#pragma unroll
    for (int i = 0; i < 4; ++i) {
      const unsigned short* p = &tA[wmBase + i * 16 + l15][khalf * 8];
      af[i].q[0] = *(const uint4*)p;
      af[i].q[1] = *(const uint4*)(p + 16);
    }
#pragma unroll
    for (int j = 0; j < 4; ++j) {
      const unsigned short* p = &tB[wnBase + j * 16 + l15][khalf * 8];
      bfr[j].q[0] = *(const uint4*)p;
      bfr[j].q[1] = *(const uint4*)(p + 16);
    }
#pragma unroll
    for (int i = 0; i < 4; ++i)
#pragma unroll
      for (int j = 0; j < 4; ++j)
        acc[i][j] = __builtin_amdgcn_wmma_f32_16x16x32_bf16(
            false, af[i].v, false, bfr[j].v, (short)0, acc[i][j], false, false);
    __syncthreads(); // done reading buf before TDM may overwrite it
  }

  // epilogue: C frag layout -> row = base + r + 8*khalf, col = base + (lane&15)
#pragma unroll
  for (int fi = 0; fi < 4; ++fi) {
    const int rowb = bm + wmBase + fi * 16 + khalf * 8;
#pragma unroll
    for (int fj = 0; fj < 4; ++fj) {
      const int col = bn + wnBase + fj * 16 + l15;
      if (col >= N) continue;
#pragma unroll
      for (int r = 0; r < 8; ++r) {
        float v = acc[fi][fj][r];
        const size_t row = (size_t)(rowb + r);
        if (EPI == 1) {
          v += bias[col];
          v = (v > 20.0f) ? v : log1pf(__expf(v)); // softplus
        } else if (EPI == 2) {
          v += resid[row * ldr + col];
        }
        C[row * ldc + col] = v;
        if (EPI == 2 || EPI == 3) Cbf[row * ldc + col] = bf16_of(v);
      }
    }
  }
}

// ---------------------------------------------------------------------------
// f32 -> bf16 bulk convert (8 elems/thread, packed v_perm stores)
// ---------------------------------------------------------------------------
__global__ __launch_bounds__(256) void convert_bf16_kernel(
    const float* __restrict__ src, unsigned int* __restrict__ dst) {
  const size_t i = ((size_t)blockIdx.x * 256 + threadIdx.x) * 8;
  const float4 f0 = *(const float4*)(src + i);
  const float4 f1 = *(const float4*)(src + i + 4);
  uint4 o;
  o.x = pack_bf16(f0.x, f0.y);
  o.y = pack_bf16(f0.z, f0.w);
  o.z = pack_bf16(f1.x, f1.y);
  o.w = pack_bf16(f1.z, f1.w);
  *(uint4*)((unsigned short*)dst + i) = o;
}

// ---------------------------------------------------------------------------
// Depthwise causal conv (4 taps) + bias + SiLU; emits f32 u and bf16 u
// ---------------------------------------------------------------------------
__global__ __launch_bounds__(256) void conv_silu_kernel(
    const float* __restrict__ xz, const float* __restrict__ cw,
    const float* __restrict__ cb, float* __restrict__ u,
    unsigned short* __restrict__ ubf) {
  const int idx = blockIdx.x * 256 + threadIdx.x; // (b*SEQ + l)*D_INNER + d
  const int d = idx & (D_INNER - 1);
  const int l = (idx >> 11) & (SEQ - 1);
  const int b = idx >> 21;
  float s = cb[d];
#pragma unroll
  for (int j = 0; j < D_CONV; ++j) {
    const int ls = l + j - (D_CONV - 1);
    if (ls >= 0)
      s += cw[d * D_CONV + j] * xz[(size_t)(b * SEQ + ls) * (2 * D_INNER) + d];
  }
  s = s / (1.0f + __expf(-s));
  u[(size_t)idx] = s;
  ubf[(size_t)idx] = bf16_of(s);
}

// ---------------------------------------------------------------------------
// Selective scan: thread per (b,d); h[16] in registers; B/C staged in LDS.
// Fuses D-skip and SiLU(z) gate; writes y directly as bf16 for out_proj.
// ---------------------------------------------------------------------------
__global__ __launch_bounds__(256) void scan_kernel(
    const float* __restrict__ dtb, const float* __restrict__ u,
    const float* __restrict__ xdbl, const float* __restrict__ xz,
    const float* __restrict__ A_log, const float* __restrict__ Dp,
    unsigned short* __restrict__ ybf) {
  const int b = blockIdx.x;
  const int d = blockIdx.y * 256 + threadIdx.x;
  __shared__ float sBC[2 * D_STATE];

  float Ac[D_STATE];
#pragma unroll
  for (int s = 0; s < D_STATE; ++s) Ac[s] = -__expf(A_log[d * D_STATE + s]);
  const float dskip = Dp[d];
  float h[D_STATE];
#pragma unroll
  for (int s = 0; s < D_STATE; ++s) h[s] = 0.0f;

  for (int l = 0; l < SEQ; ++l) {
    const size_t m = (size_t)b * SEQ + l;
    if (threadIdx.x < 2 * D_STATE)
      sBC[threadIdx.x] = xdbl[m * XDBL_W + DT_RANK + threadIdx.x];
    __syncthreads();

    const float dtv = dtb[m * D_INNER + d];
    const float uv = u[m * D_INNER + d];
    const float du = dtv * uv;
    float acc = 0.0f;
#pragma unroll
    for (int s = 0; s < D_STATE; ++s) {
      const float dA = __expf(dtv * Ac[s]);
      h[s] = dA * h[s] + du * sBC[s];
      acc += h[s] * sBC[D_STATE + s];
    }
    acc += uv * dskip;
    const float z = xz[m * (2 * D_INNER) + D_INNER + d];
    acc *= z / (1.0f + __expf(-z));
    ybf[m * D_INNER + d] = bf16_of(acc);
    __syncthreads();
  }
}

// ---------------------------------------------------------------------------
extern "C" void kernel_launch(void* const* d_in, const int* in_sizes, int n_in,
                              void* d_out, int out_size, void* d_ws,
                              size_t ws_size, hipStream_t stream) {
  (void)in_sizes; (void)n_in; (void)out_size; (void)ws_size;
  const float* x_in = (const float*)d_in[0];
  const float* wi_all = (const float*)d_in[1];
  const float* cw_all = (const float*)d_in[2];
  const float* cb_all = (const float*)d_in[3];
  const float* wx_all = (const float*)d_in[4];
  const float* wdt_all = (const float*)d_in[5];
  const float* bdt_all = (const float*)d_in[6];
  const float* alog_all = (const float*)d_in[7];
  const float* dp_all = (const float*)d_in[8];
  const float* wo_all = (const float*)d_in[9];

  char* ws = (char*)d_ws;
  float* xz = (float*)ws;   ws += (size_t)MROWS * 2 * D_INNER * 4;
  float* u = (float*)ws;    ws += (size_t)MROWS * D_INNER * 4;
  float* xdbl = (float*)ws; ws += (size_t)MROWS * XDBL_W * 4;
  float* dtb = (float*)ws;  ws += (size_t)MROWS * D_INNER * 4;
  float* xb = (float*)ws;   ws += (size_t)MROWS * D_MODEL * 4;
  unsigned short* ubf = (unsigned short*)ws;    ws += (size_t)MROWS * D_INNER * 2;
  unsigned short* xdblbf = (unsigned short*)ws; ws += (size_t)MROWS * XDBL_W * 2;
  unsigned short* ybf = (unsigned short*)ws;    ws += (size_t)MROWS * D_INNER * 2;
  unsigned short* xbf = (unsigned short*)ws;    ws += (size_t)MROWS * D_MODEL * 2;
  unsigned short* wibf = (unsigned short*)ws;   ws += (size_t)2 * D_INNER * D_MODEL * 2;
  unsigned short* wxbf = (unsigned short*)ws;   ws += (size_t)XDBL_W * D_INNER * 2;
  unsigned short* wdtbf = (unsigned short*)ws;  ws += (size_t)D_INNER * DT_RANK * 2;
  unsigned short* wobf = (unsigned short*)ws;   ws += (size_t)D_MODEL * D_INNER * 2;

  // bf16 image of the input activations (later layers get it fused from EPI 2)
  convert_bf16_kernel<<<(MROWS * D_MODEL) / 2048, 256, 0, stream>>>(
      x_in, (unsigned int*)xbf);

  for (int layer = 0; layer < DEPTH; ++layer) {
    const float* xcur = (layer == 0) ? x_in : xb;
    float* xnext = (layer == DEPTH - 1) ? (float*)d_out : xb;
    const float* wi = wi_all + (size_t)layer * 2 * D_INNER * D_MODEL;
    const float* cw = cw_all + (size_t)layer * D_INNER * D_CONV;
    const float* cb = cb_all + (size_t)layer * D_INNER;
    const float* wx = wx_all + (size_t)layer * XDBL_W * D_INNER;
    const float* wdt = wdt_all + (size_t)layer * D_INNER * DT_RANK;
    const float* bdt = bdt_all + (size_t)layer * D_INNER;
    const float* alog = alog_all + (size_t)layer * D_INNER * D_STATE;
    const float* dp = dp_all + (size_t)layer * D_INNER;
    const float* wo = wo_all + (size_t)layer * D_MODEL * D_INNER;

    // per-layer weight slices -> bf16
    convert_bf16_kernel<<<(2 * D_INNER * D_MODEL) / 2048, 256, 0, stream>>>(
        wi, (unsigned int*)wibf);
    convert_bf16_kernel<<<(XDBL_W * D_INNER) / 2048, 256, 0, stream>>>(
        wx, (unsigned int*)wxbf);
    convert_bf16_kernel<<<(D_INNER * DT_RANK) / 2048, 256, 0, stream>>>(
        wdt, (unsigned int*)wdtbf);
    convert_bf16_kernel<<<(D_MODEL * D_INNER) / 2048, 256, 0, stream>>>(
        wo, (unsigned int*)wobf);

    // in_proj: xz[2048,4096] = x * Wi^T
    gemm_wmma_bf16<0><<<dim3(MROWS / BM, (2 * D_INNER) / BN), 256, 0, stream>>>(
        xbf, D_MODEL, wibf, D_MODEL, xz, 2 * D_INNER, nullptr, nullptr,
        nullptr, 0, MROWS, 2 * D_INNER, D_MODEL);

    // depthwise conv + silu -> u (f32 + bf16)
    conv_silu_kernel<<<(BATCH * SEQ * D_INNER) / 256, 256, 0, stream>>>(
        xz, cw, cb, u, ubf);

    // x_proj: xdbl[2048,96] = u * Wx^T   (f32 for scan + bf16 for dt GEMM)
    gemm_wmma_bf16<3><<<dim3(MROWS / BM, 1), 256, 0, stream>>>(
        ubf, D_INNER, wxbf, D_INNER, xdbl, XDBL_W, xdblbf, nullptr, nullptr, 0,
        MROWS, XDBL_W, D_INNER);

    // dt_proj + softplus: dtb = softplus(xdbl[:, :64] * Wdt^T + bdt)
    gemm_wmma_bf16<1><<<dim3(MROWS / BM, D_INNER / BN), 256, 0, stream>>>(
        xdblbf, XDBL_W, wdtbf, DT_RANK, dtb, D_INNER, nullptr, bdt, nullptr, 0,
        MROWS, D_INNER, DT_RANK);

    // selective scan + D skip + silu(z) gate -> ybf (bf16)
    scan_kernel<<<dim3(BATCH, D_INNER / 256), 256, 0, stream>>>(
        dtb, u, xdbl, xz, alog, dp, ybf);

    // out_proj + residual: xnext = y * Wo^T + xcur (f32), plus bf16 x for next
    gemm_wmma_bf16<2><<<dim3(MROWS / BM, D_MODEL / BN), 256, 0, stream>>>(
        ybf, D_INNER, wobf, D_INNER, xnext, D_MODEL, xbf, nullptr, xcur,
        D_MODEL, MROWS, D_MODEL, D_INNER);
  }
}
